// RNNBlock_7215545057997
// MI455X (gfx1250) — compile-verified, matrix operand order checked by reading, not run on hardware
//
#include <hip/hip_runtime.h>
#include <stddef.h>
#include <stdint.h>

// ---------------------------------------------------------------------------
// Bidirectional GRU block for MI455X (gfx1250, wave32, WMMA).
//   B=32, T=1000, I=H=512, G=3H=1536, M=T*B=32000
// Pipeline:
//   1) zero_kernel          : clear barrier + h bf16 ping-pong state in ws
//   2) ln_relu6_kernel      : LayerNorm+ReLU6, transpose to [T,B,I], f32->bf16
//   3) cvt_bf16_kernel x4   : weights f32->bf16
//   4) gemm_xg_kernel       : xg = xt @ W_ih^T + b_ih  (bf16 WMMA, f32 acc/out)
//   5) gru_scan_kernel      : persistent, 16 blocks (8/dir);
//                             w_hh slice staged ONCE into LDS via TDM
//                             (tensor_load_to_lds, 3 x 64KB 2D descriptors);
//                             h staged per step via global_load_async_to_lds;
//                             f32 h carry lives in VGPRs; 1 grid barrier/step.
// ---------------------------------------------------------------------------

#define BB 32
#define TT 1000
#define II 512
#define HH 512
#define GG 1536
#define MM (TT * BB)

// workspace layout (bytes, all 256-aligned)
#define OFF_BAR   ((size_t)0)                    // 256 B barrier area
#define OFF_HB    ((size_t)256)                  // h bf16 [2 dir][2 buf][32][512]
#define OFF_XT    ((size_t)131328)               // xt bf16 [T*B][512]
#define OFF_WIHF  ((size_t)32899328)             // w_ih_f bf16 [1536][512]
#define OFF_WIHB  ((size_t)34472192)
#define OFF_WHHF  ((size_t)36045056)
#define OFF_WHHB  ((size_t)37617920)
#define OFF_XGF   ((size_t)39190784)             // xg_f f32 [T*B][1536]
#define OFF_XGB   ((size_t)235798784)
#define ZERO_WORDS ((int)((256 + 131072) / 4))

// scan-kernel LDS: [0,196608) = w_hh slice (192 rows x 512 bf16)
//                  [196608, 229376) = h stage (32 x 512 bf16)
#define SCAN_LDS_BYTES 229376
#define WLDS_ELEMS 98304   // 192*512

typedef __attribute__((ext_vector_type(16))) __bf16 v16bf;
typedef __attribute__((ext_vector_type(8)))  float  v8f;
typedef __attribute__((ext_vector_type(4)))  unsigned su4;   // SGPR quad
typedef __attribute__((ext_vector_type(8)))  unsigned su8;   // SGPR octet

union Frag16 { v16bf v; uint4 u[2]; };

__device__ __forceinline__ unsigned short f32_to_bf16(float f) {
  unsigned u = __float_as_uint(f);
  unsigned r = u + 0x7FFFu + ((u >> 16) & 1u);   // round-to-nearest-even
  return (unsigned short)(r >> 16);
}

// Low 32 bits of a generic pointer to LDS == byte offset usable by DS/async ops
__device__ __forceinline__ unsigned lds_off(const void* p) {
  return (unsigned)(size_t)p;
}

// CDNA5 async copy: 16 bytes per active lane, global -> LDS, ASYNCcnt-tracked.
__device__ __forceinline__ void async_copy_b128(unsigned lds_addr, const void* gaddr) {
  asm volatile("global_load_async_to_lds_b128 %0, %1, off"
               :
               : "v"(lds_addr), "v"((unsigned long long)(size_t)gaddr)
               : "memory");
}
__device__ __forceinline__ void wait_async0() {
  asm volatile("s_wait_asynccnt 0x0" ::: "memory");
}

// Tensor Data Mover: 2D tile (tile_d1 rows x tile_d0 elements, 2-byte data)
// from global (row stride row_stride elems) into contiguous LDS at lds_addr.
// D# built per CDNA5 ISA; 2-group form (<=2D).
__device__ __forceinline__ void tdm_load_2d_bf16(unsigned lds_addr,
                                                 const void* gaddr,
                                                 unsigned tensor_d0,
                                                 unsigned tensor_d1,
                                                 unsigned tile_d0,
                                                 unsigned tile_d1,
                                                 unsigned row_stride) {
  const unsigned long long ga = (unsigned long long)(size_t)gaddr;
  su4 g0;
  g0[0] = 1u;                                            // count=1 (valid D#)
  g0[1] = lds_addr;                                      // lds_addr[31:0]
  g0[2] = (unsigned)(ga & 0xFFFFFFFFu);                  // global_addr[31:0]
  g0[3] = (unsigned)((ga >> 32) & 0x01FFFFFFu)           // global_addr[56:32]
          | 0x80000000u;                                 // type=2 ("image")
  su8 g1;
  g1[0] = (1u << 16);                                    // data_size=1 (2B)
  g1[1] = (tensor_d0 & 0xFFFFu) << 16;                   // tensor_dim0 lo16
  g1[2] = ((tensor_d0 >> 16) & 0xFFFFu) | ((tensor_d1 & 0xFFFFu) << 16);
  g1[3] = ((tensor_d1 >> 16) & 0xFFFFu) | ((tile_d0 & 0xFFFFu) << 16);
  g1[4] = (tile_d1 & 0xFFFFu);                           // tile_dim1, tile_dim2=0
  g1[5] = row_stride;                                    // tensor_dim0_stride lo32
  g1[6] = 0u;                                            // stride hi16 | dim1_stride lo16
  g1[7] = 0u;
  asm volatile("tensor_load_to_lds %0, %1"
               :
               : "s"(g0), "s"(g1)
               : "memory");
}

// Load a 16x32 bf16 tile (rows along M/N, 32 along K) from row-major storage
// into the CDNA5 WMMA A/B fragment layout:
//   lanes 0-15 : row = lane,    K = {0..7, 16..23}
//   lanes 16-31: row = lane-16, K = {8..15, 24..31}
// Works for both global and LDS (address-space inferred -> ds_load_b128).
__device__ __forceinline__ v16bf load_tile(const unsigned short* __restrict__ colbase,
                                           int row0, int ld) {
  const int lane = threadIdx.x & 31;
  const size_t r = (size_t)(row0 + (lane & 15)) * (size_t)ld;
  const int k0 = (lane >> 4) * 8;
  Frag16 f;
  f.u[0] = *(const uint4*)(colbase + r + k0);
  f.u[1] = *(const uint4*)(colbase + r + k0 + 16);
  return f.v;
}

__device__ __forceinline__ v8f wmma_bf16(v16bf a, v16bf b, v8f c) {
  return __builtin_amdgcn_wmma_f32_16x16x32_bf16(
      false, a, false, b, (short)0, c, false, false);
}

// ---------------------------------------------------------------------------
__global__ void zero_kernel(unsigned* __restrict__ p, int n) {
  int i = blockIdx.x * blockDim.x + threadIdx.x;
  if (i < n) p[i] = 0u;
}

__global__ void cvt_bf16_kernel(const float* __restrict__ s,
                                unsigned short* __restrict__ d, int n) {
  int i = blockIdx.x * blockDim.x + threadIdx.x;
  if (i < n) d[i] = f32_to_bf16(s[i]);
}

// ---------------------------------------------------------------------------
// LayerNorm + ReLU6 + transpose [B,T,I] -> [T,B,I], output bf16.
__global__ void ln_relu6_kernel(const float* __restrict__ x,
                                const float* __restrict__ gamma,
                                const float* __restrict__ beta,
                                unsigned short* __restrict__ xt) {
  const int row = blockIdx.x;            // row = b*T + t
  const int b = row / TT;
  const int t = row - b * TT;
  const float* xr = x + (size_t)row * II;
  const int tid = threadIdx.x;

  float v0 = xr[tid];
  float v1 = xr[tid + 256];
  float s = v0 + v1;
  float s2 = v0 * v0 + v1 * v1;
#pragma unroll
  for (int o = 16; o > 0; o >>= 1) {
    s  += __shfl_xor(s, o, 32);
    s2 += __shfl_xor(s2, o, 32);
  }
  __shared__ float sh[16];
  const int w = tid >> 5, l = tid & 31;
  if (l == 0) { sh[w] = s; sh[8 + w] = s2; }
  __syncthreads();
  if (w == 0) {
    float a  = (l < 8) ? sh[l] : 0.f;
    float a2 = (l < 8) ? sh[8 + l] : 0.f;
#pragma unroll
    for (int o = 4; o > 0; o >>= 1) {
      a  += __shfl_xor(a, o, 32);
      a2 += __shfl_xor(a2, o, 32);
    }
    if (l == 0) { sh[0] = a; sh[1] = a2; }
  }
  __syncthreads();
  const float mean = sh[0] * (1.0f / II);
  const float var  = sh[1] * (1.0f / II) - mean * mean;
  const float inv  = rsqrtf(var + 1e-5f);

  float y0 = (v0 - mean) * inv * gamma[tid]       + beta[tid];
  float y1 = (v1 - mean) * inv * gamma[tid + 256] + beta[tid + 256];
  y0 = fminf(fmaxf(y0, 0.0f), 6.0f);
  y1 = fminf(fmaxf(y1, 0.0f), 6.0f);

  const size_t ob = ((size_t)t * BB + b) * II;
  xt[ob + tid]       = f32_to_bf16(y0);
  xt[ob + tid + 256] = f32_to_bf16(y1);
}

// ---------------------------------------------------------------------------
// xg = xt(bf16)[M,512] @ W(bf16)[1536,512]^T + b  -> f32 [M,1536]
// Block = 8 waves (4 along M x 2 along N); wave tile 32x64; block tile 128x128.
// Direct-from-L2: xt (32MB) and both weight matrices stay resident in the
// 192MB L2; round-1 asm confirmed a fully pipelined clause/wmma loop.
__global__ void __launch_bounds__(256)
gemm_xg_kernel(const unsigned short* __restrict__ xt,
               const unsigned short* __restrict__ w_f,
               const unsigned short* __restrict__ w_b,
               const float* __restrict__ bias_f,
               const float* __restrict__ bias_b,
               float* __restrict__ xg_f,
               float* __restrict__ xg_b) {
  const int dir = blockIdx.z;
  const unsigned short* w = dir ? w_b : w_f;
  const float* bias = dir ? bias_b : bias_f;
  float* xg = dir ? xg_b : xg_f;

  const int wave = threadIdx.x >> 5;
  const int wm = wave & 3;
  const int wn = wave >> 2;
  const int m0 = blockIdx.y * 128 + wm * 32;
  const int n0 = blockIdx.x * 128 + wn * 64;

  v8f acc[2][4] = {};
#pragma unroll 4
  for (int kk = 0; kk < 16; ++kk) {
    const unsigned short* xc = xt + kk * 32;
    const unsigned short* wc = w + kk * 32;
    v16bf a0 = load_tile(xc, m0, II);
    v16bf a1 = load_tile(xc, m0 + 16, II);
#pragma unroll
    for (int nt = 0; nt < 4; ++nt) {
      v16bf bt = load_tile(wc, n0 + nt * 16, II);
      acc[0][nt] = wmma_bf16(a0, bt, acc[0][nt]);
      acc[1][nt] = wmma_bf16(a1, bt, acc[1][nt]);
    }
  }

  const int lane = threadIdx.x & 31;
  const int nl = lane & 15;
  const int mh = (lane >> 4) * 8;
#pragma unroll
  for (int nt = 0; nt < 4; ++nt) {
    const int col = n0 + nt * 16 + nl;
    const float bv = bias[col];
#pragma unroll
    for (int mt = 0; mt < 2; ++mt) {
#pragma unroll
      for (int i = 0; i < 8; ++i) {
        const int m = m0 + mt * 16 + mh + i;
        xg[(size_t)m * GG + col] = acc[mt][nt][i] + bv;
      }
    }
  }
}

// ---------------------------------------------------------------------------
// grid barrier among nblk cooperating blocks; bar2[0]=count, bar2[1]=generation
__device__ __forceinline__ void group_barrier(unsigned* bar2, unsigned nblk) {
  __syncthreads();
  __threadfence();   // all waves release their stores to agent scope
  if (threadIdx.x == 0) {
    unsigned gen = __hip_atomic_load(&bar2[1], __ATOMIC_RELAXED,
                                     __HIP_MEMORY_SCOPE_AGENT);
    unsigned ticket = __hip_atomic_fetch_add(&bar2[0], 1u, __ATOMIC_ACQ_REL,
                                             __HIP_MEMORY_SCOPE_AGENT);
    if (ticket == nblk - 1) {
      __hip_atomic_store(&bar2[0], 0u, __ATOMIC_RELAXED,
                         __HIP_MEMORY_SCOPE_AGENT);
      __hip_atomic_fetch_add(&bar2[1], 1u, __ATOMIC_RELEASE,
                             __HIP_MEMORY_SCOPE_AGENT);
    } else {
      while (__hip_atomic_load(&bar2[1], __ATOMIC_ACQUIRE,
                               __HIP_MEMORY_SCOPE_AGENT) == gen) {
        __builtin_amdgcn_s_sleep(1);
      }
    }
  }
  __syncthreads();
  // every wave (both CUs of the WGP) drops stale lines before re-reading
  __builtin_amdgcn_fence(__ATOMIC_ACQUIRE, "agent");
}

// ---------------------------------------------------------------------------
// Persistent bidirectional GRU scan.
// 16 blocks: 0-7 forward, 8-15 backward.  Each block owns a 64-wide h-column
// slice; its w_hh rows {j, H+j, 2H+j} (192x512 bf16 = 192KB) are staged ONCE
// into LDS via three TDM descriptors and stay resident for all 1000 steps.
// The shared 32KB h state is staged into LDS per step via per-lane async
// copies.  The f32 h carry lives in VGPRs (accumulator layout); only the bf16
// h copy round-trips global memory for the cross-block handoff (ping-pong,
// one grid barrier per step).
__global__ void __launch_bounds__(128)
gru_scan_kernel(const unsigned short* __restrict__ whh_f,
                const unsigned short* __restrict__ whh_b,
                const float* __restrict__ bhh_f,
                const float* __restrict__ bhh_b,
                const float* __restrict__ xg_f,
                const float* __restrict__ xg_b,
                unsigned short* __restrict__ hbf,
                unsigned* __restrict__ bar,
                float* __restrict__ out) {
  extern __shared__ unsigned short smem[];
  unsigned short* w_lds = smem;                // [192][512] bf16
  unsigned short* h_lds = smem + WLDS_ELEMS;   // [32][512] bf16

  const int dir   = blockIdx.x >> 3;
  const int slice = blockIdx.x & 7;
  const int tid   = threadIdx.x;
  const int wave  = tid >> 5;
  const int lane  = tid & 31;
  const int jw    = slice * 64 + wave * 16;    // this wave's h-column base

  const unsigned short* whh = dir ? whh_b : whh_f;
  const float* bhh = dir ? bhh_b : bhh_f;
  const float* xg  = dir ? xg_b : xg_f;
  unsigned* mybar  = bar + dir * 16;           // 64B-separated per direction
  unsigned short* hb_base = hbf + (size_t)dir * 2 * BB * HH;

  const int nl = lane & 15;
  const int mh = (lane >> 4) * 8;
  const int colh = jw + nl;

  // ---- one-time: stage this block's 192 w_hh rows into LDS via TDM ----
  // One 2D descriptor per gate set: 64 rows x 512 elems (64KB), row stride
  // 512; destination is contiguous in LDS => [s*64 + r][k] layout.
  if (wave == 0) {
    const unsigned wbase = lds_off(w_lds);
#pragma unroll
    for (int s = 0; s < 3; ++s) {
      tdm_load_2d_bf16(wbase + (unsigned)(s * 64 * II * 2),
                       whh + (size_t)(s * HH + slice * 64) * II,
                       /*tensor_d0=*/II, /*tensor_d1=*/GG,
                       /*tile_d0=*/II, /*tile_d1=*/64,
                       /*row_stride=*/II);
    }
    __builtin_amdgcn_s_wait_tensorcnt(0);
  }
  __syncthreads();

  // biases are step-invariant
  const float bh0 = bhh[0 * HH + colh];
  const float bh1 = bhh[1 * HH + colh];
  const float bh2 = bhh[2 * HH + colh];

  // f32 h carry in registers, accumulator layout: [mt][i] -> (m = mt*16+mh+i)
  float hcarry[2][8] = {};

  const unsigned hbase_lds = lds_off(h_lds);

  for (int step = 0; step < TT; ++step) {
    const int rb = step & 1, wb = rb ^ 1;
    const int tt = dir ? (TT - 1 - step) : step;
    const unsigned short* hread = hb_base + (size_t)rb * BB * HH;

    // stage shared h state (32x512 bf16 = 32KB) into LDS: 16 x 2KB async
#pragma unroll
    for (int it = 0; it < 16; ++it) {
      const int e = (it * 128 + tid) * 8;
      async_copy_b128(hbase_lds + (unsigned)(e * 2), hread + e);
    }

    // prefetch next step's xg slab while the DMA runs
    if (step + 1 < TT) {
      const int tn = dir ? (TT - 2 - step) : (step + 1);
      __builtin_prefetch(xg + ((size_t)tn * BB) * GG + (size_t)lane * 16, 0, 1);
    }

    wait_async0();
    __syncthreads();

    v8f acc[2][3] = {};
#pragma unroll 4
    for (int kk = 0; kk < 16; ++kk) {
      const unsigned short* hc = h_lds + kk * 32;
      const unsigned short* wc = w_lds + kk * 32;
      v16bf a0 = load_tile(hc, 0, HH);
      v16bf a1 = load_tile(hc, 16, HH);
#pragma unroll
      for (int s = 0; s < 3; ++s) {
        v16bf bt = load_tile(wc, s * 64 + wave * 16, HH);
        acc[0][s] = wmma_bf16(a0, bt, acc[0][s]);
        acc[1][s] = wmma_bf16(a1, bt, acc[1][s]);
      }
    }

    unsigned short* hnew_b = hb_base + (size_t)wb * BB * HH;

#pragma unroll
    for (int mt = 0; mt < 2; ++mt) {
#pragma unroll
      for (int i = 0; i < 8; ++i) {
        const int m = mt * 16 + mh + i;        // batch index
        const size_t xrow = ((size_t)tt * BB + m) * GG;
        const float xr_ = xg[xrow + 0 * HH + colh];
        const float xz_ = xg[xrow + 1 * HH + colh];
        const float xn_ = xg[xrow + 2 * HH + colh];
        const float hgr = acc[mt][0][i] + bh0;
        const float hgz = acc[mt][1][i] + bh1;
        const float hgn = acc[mt][2][i] + bh2;
        const float r = 1.0f / (1.0f + __expf(-(xr_ + hgr)));
        const float z = 1.0f / (1.0f + __expf(-(xz_ + hgz)));
        const float nn = tanhf(xn_ + r * hgn);
        const float hv = (1.0f - z) * nn + z * hcarry[mt][i];
        hcarry[mt][i] = hv;
        hnew_b[(size_t)m * HH + colh] = f32_to_bf16(hv);
        out[((size_t)m * TT + tt) * (2 * HH) + dir * HH + colh] = hv;
      }
    }

    group_barrier(mybar, 8u);
  }
}

// ---------------------------------------------------------------------------
extern "C" void kernel_launch(void* const* d_in, const int* in_sizes, int n_in,
                              void* d_out, int out_size, void* d_ws,
                              size_t ws_size, hipStream_t stream) {
  (void)in_sizes; (void)n_in; (void)out_size; (void)ws_size;
  const float* x      = (const float*)d_in[0];
  const float* ln_g   = (const float*)d_in[1];
  const float* ln_b   = (const float*)d_in[2];
  const float* w_ih_f = (const float*)d_in[3];
  const float* w_hh_f = (const float*)d_in[4];
  const float* b_ih_f = (const float*)d_in[5];
  const float* b_hh_f = (const float*)d_in[6];
  const float* w_ih_b = (const float*)d_in[7];
  const float* w_hh_b = (const float*)d_in[8];
  const float* b_ih_b = (const float*)d_in[9];
  const float* b_hh_b = (const float*)d_in[10];

  char* ws = (char*)d_ws;
  unsigned*       bar  = (unsigned*)(ws + OFF_BAR);
  unsigned short* hb   = (unsigned short*)(ws + OFF_HB);
  unsigned short* xt   = (unsigned short*)(ws + OFF_XT);
  unsigned short* wihf = (unsigned short*)(ws + OFF_WIHF);
  unsigned short* wihb = (unsigned short*)(ws + OFF_WIHB);
  unsigned short* whhf = (unsigned short*)(ws + OFF_WHHF);
  unsigned short* whhb = (unsigned short*)(ws + OFF_WHHB);
  float*          xgf  = (float*)(ws + OFF_XGF);
  float*          xgb  = (float*)(ws + OFF_XGB);

  zero_kernel<<<(ZERO_WORDS + 255) / 256, 256, 0, stream>>>((unsigned*)ws,
                                                            ZERO_WORDS);
  ln_relu6_kernel<<<MM, 256, 0, stream>>>(x, ln_g, ln_b, xt);

  const int WN = GG * II;  // 786432
  cvt_bf16_kernel<<<(WN + 255) / 256, 256, 0, stream>>>(w_ih_f, wihf, WN);
  cvt_bf16_kernel<<<(WN + 255) / 256, 256, 0, stream>>>(w_ih_b, wihb, WN);
  cvt_bf16_kernel<<<(WN + 255) / 256, 256, 0, stream>>>(w_hh_f, whhf, WN);
  cvt_bf16_kernel<<<(WN + 255) / 256, 256, 0, stream>>>(w_hh_b, whhb, WN);

  dim3 gg(GG / 128, MM / 128, 2);
  gemm_xg_kernel<<<gg, 256, 0, stream>>>(xt, wihf, wihb, b_ih_f, b_ih_b, xgf,
                                         xgb);

  gru_scan_kernel<<<16, 128, SCAN_LDS_BYTES, stream>>>(
      whhf, whhb, b_hh_f, b_hh_b, xgf, xgb, hb, bar, (float*)d_out);
}
